// GGNNWithEdgeTypes_84129819394286
// MI455X (gfx1250) — matrix operand order repaired and checked by reading, not compiled, not run on hardware
//
#include <hip/hip_runtime.h>

#define NN     50000
#define EE     800000
#define DIN    369
#define DINP   384
#define DD     128
#define TT     13
#define NSTEPS 8
#define GG     256
#define MT     (NN / 16)   // 3125 row tiles, exact

typedef __attribute__((ext_vector_type(16))) __bf16         v16bf;
typedef __attribute__((ext_vector_type(8)))  float          v8f;
typedef __attribute__((ext_vector_type(8)))  unsigned short ush8;

union BF16x16 { v16bf v; unsigned short u[16]; ush8 h[2]; };

__device__ __forceinline__ unsigned short f2bf(float f) {
  unsigned int u = __float_as_uint(f);
  unsigned int r = u + 0x7FFFu + ((u >> 16) & 1u);   // RNE to bf16
  return (unsigned short)(r >> 16);
}

__device__ __forceinline__ v8f wmma_bf16(v16bf a, v16bf b, v8f c) {
  return __builtin_amdgcn_wmma_f32_16x16x32_bf16(false, a, false, b, (short)0, c,
                                                 false, false);
}

__device__ __forceinline__ float sigm(float x) { return 1.0f / (1.0f + __expf(-x)); }

// ---------------- utility kernels ----------------

__global__ void k_zero(float* p, long long n4) {
  long long i = (long long)blockIdx.x * blockDim.x + threadIdx.x;
  if (i < n4) {
    float4 z; z.x = z.y = z.z = z.w = 0.0f;
    ((float4*)p)[i] = z;
  }
}

__global__ void k_cvt_x(const float* __restrict__ x, unsigned short* __restrict__ o) {
  long long i = (long long)blockIdx.x * blockDim.x + threadIdx.x;
  if (i >= (long long)NN * DINP) return;
  int n = (int)(i / DINP), k = (int)(i % DINP);
  o[i] = f2bf(k < DIN ? x[(long long)n * DIN + k] : 0.0f);
}

__global__ void k_cvt_linw(const float* __restrict__ lw, unsigned short* __restrict__ o) {
  int i = blockIdx.x * blockDim.x + threadIdx.x;
  if (i >= DD * DINP) return;
  int n = i / DINP, k = i % DINP;               // column-major: [n][k]
  o[i] = f2bf(k < DIN ? lw[k * DD + n] : 0.0f);
}

__global__ void k_cvt_bf(const float* __restrict__ in, unsigned short* __restrict__ o,
                         long long n) {
  long long i = (long long)blockIdx.x * blockDim.x + threadIdx.x;
  if (i < n) o[i] = f2bf(in[i]);
}

// Wc[t][s][n][k] = sum_p ggnn_w[t][s][k][p] * w_ih[t][n][p]   (bf16 out, col-major B)
__global__ void k_wc(const float* __restrict__ ggnn, const float* __restrict__ wih,
                     unsigned short* __restrict__ wc) {
  int i = blockIdx.x * blockDim.x + threadIdx.x;
  if (i >= TT * NSTEPS * 384 * DD) return;
  int k = i & 127;
  int n = (i >> 7) % 384;
  int ts = i / (384 * DD);
  const float* ga = ggnn + ((long long)ts * DD + k) * DD;
  const float* wi = wih + ((long long)(ts / NSTEPS) * 384 + n) * DD;
  float acc = 0.0f;
  for (int p = 0; p < DD; p += 4) {
    float4 a = *(const float4*)(ga + p);
    float4 b = *(const float4*)(wi + p);
    acc += a.x * b.x + a.y * b.y + a.z * b.z + a.w * b.w;
  }
  wc[i] = f2bf(acc);
}

// ---------------- h0 = x @ lin_w + lin_b, broadcast to 13 H slices ----------------

__global__ __launch_bounds__(256) void k_h0(const unsigned short* __restrict__ xbf,
                                            const unsigned short* __restrict__ lwcm,
                                            const float* __restrict__ lin_b,
                                            float* __restrict__ H) {
  int gw = blockIdx.x * 8 + (threadIdx.x >> 5);
  if (gw >= MT * 8) return;
  int lane = threadIdx.x & 31;
  int mt = gw >> 3, j = gw & 7;
  int m0 = mt * 16, l15 = lane & 15, lh = lane >> 4;
  const unsigned short* arow = xbf + (long long)(m0 + l15) * DINP;
  const unsigned short* brow = lwcm + (long long)(j * 16 + l15) * DINP;
  v8f acc = {0.f, 0.f, 0.f, 0.f, 0.f, 0.f, 0.f, 0.f};
#pragma unroll
  for (int kc = 0; kc < DINP / 32; ++kc) {
    int kb = kc * 32 + lh * 16;
    BF16x16 a;
    a.h[0] = *(const ush8*)(arow + kb);
    a.h[1] = *(const ush8*)(arow + kb + 8);
    v16bf b = *(const v16bf*)(brow + kc * 32 + lh * 16);
    acc = wmma_bf16(a.v, b, acc);
  }
  int col = j * 16 + l15;
  float bias = lin_b[col];
  int rbase = m0 + lh * 8;
#pragma unroll
  for (int v = 0; v < 8; ++v) {
    float val = acc[v] + bias;
#pragma unroll
    for (int t = 0; t < TT; ++t)
      H[((long long)t * NN + rbase + v) * DD + col] = val;
  }
}

// ---------------- edge scatter: S[t][dst] += H[t][src] ----------------

__global__ void k_scatter(const float* __restrict__ H, float* __restrict__ S,
                          const int* __restrict__ eidx, const int* __restrict__ eattr) {
  int gw = blockIdx.x * 8 + (threadIdx.x >> 5);
  if (gw >= EE) return;
  int lane = threadIdx.x & 31;
  int t = eattr[gw];
  int sn = eidx[gw];
  int dn = eidx[EE + gw];
  const float4 v = *(const float4*)(H + ((long long)t * NN + sn) * DD + lane * 4);
  float* o = S + ((long long)t * NN + dn) * DD + lane * 4;
  atomicAdd(o + 0, v.x); atomicAdd(o + 1, v.y);
  atomicAdd(o + 2, v.z); atomicAdd(o + 3, v.w);
}

// ---------------- fused GRU step: gi = S@Wc+b_ih, gh = H@Whh+b_hh, gate, H in-place ----

__global__ __launch_bounds__(256) void k_gru(float* __restrict__ H,
                                             const float* __restrict__ S,
                                             const unsigned short* __restrict__ wc,
                                             const unsigned short* __restrict__ whh,
                                             const float* __restrict__ bih,
                                             const float* __restrict__ bhh, int s) {
  int gw = blockIdx.x * 8 + (threadIdx.x >> 5);
  if (gw >= TT * MT) return;
  int lane = threadIdx.x & 31;
  int t = gw / MT, mt = gw % MT;
  int m0 = mt * 16, l15 = lane & 15, lh = lane >> 4;

  const float* hrow = H + ((long long)t * NN + m0 + l15) * DD;
  const float* srow = S + ((long long)t * NN + m0 + l15) * DD;

  BF16x16 aH[4], aS[4];
#pragma unroll
  for (int kc = 0; kc < 4; ++kc) {
    int kb = kc * 32 + lh * 16;
#pragma unroll
    for (int q = 0; q < 2; ++q) {
      int ko = kb + q * 8;
      float4 h0v = *(const float4*)(hrow + ko);
      float4 h1v = *(const float4*)(hrow + ko + 4);
      float4 s0v = *(const float4*)(srow + ko);
      float4 s1v = *(const float4*)(srow + ko + 4);
      int b = q * 8;
      aH[kc].u[b + 0] = f2bf(h0v.x); aH[kc].u[b + 1] = f2bf(h0v.y);
      aH[kc].u[b + 2] = f2bf(h0v.z); aH[kc].u[b + 3] = f2bf(h0v.w);
      aH[kc].u[b + 4] = f2bf(h1v.x); aH[kc].u[b + 5] = f2bf(h1v.y);
      aH[kc].u[b + 6] = f2bf(h1v.z); aH[kc].u[b + 7] = f2bf(h1v.w);
      aS[kc].u[b + 0] = f2bf(s0v.x); aS[kc].u[b + 1] = f2bf(s0v.y);
      aS[kc].u[b + 2] = f2bf(s0v.z); aS[kc].u[b + 3] = f2bf(s0v.w);
      aS[kc].u[b + 4] = f2bf(s1v.x); aS[kc].u[b + 5] = f2bf(s1v.y);
      aS[kc].u[b + 6] = f2bf(s1v.z); aS[kc].u[b + 7] = f2bf(s1v.w);
    }
  }

  const unsigned short* wcb = wc + (long long)(t * NSTEPS + s) * 384 * DD;
  const unsigned short* whb = whh + (long long)t * 384 * DD;
  const float* bi = bih + t * 384;
  const float* bh = bhh + t * 384;
  const v8f z8 = {0.f, 0.f, 0.f, 0.f, 0.f, 0.f, 0.f, 0.f};

  for (int j = 0; j < 8; ++j) {
    v8f aRi = z8, aZi = z8, aNi = z8, aRh = z8, aZh = z8, aNh = z8;
    int n0 = j * 16 + l15;
#pragma unroll
    for (int kc = 0; kc < 4; ++kc) {
      int ko = kc * 32 + lh * 16;
      v16bf bRc = *(const v16bf*)(wcb + (long long)(0 * DD + n0) * DD + ko);
      v16bf bZc = *(const v16bf*)(wcb + (long long)(1 * DD + n0) * DD + ko);
      v16bf bNc = *(const v16bf*)(wcb + (long long)(2 * DD + n0) * DD + ko);
      v16bf bRh = *(const v16bf*)(whb + (long long)(0 * DD + n0) * DD + ko);
      v16bf bZh = *(const v16bf*)(whb + (long long)(1 * DD + n0) * DD + ko);
      v16bf bNh = *(const v16bf*)(whb + (long long)(2 * DD + n0) * DD + ko);
      aRi = wmma_bf16(aS[kc].v, bRc, aRi);
      aZi = wmma_bf16(aS[kc].v, bZc, aZi);
      aNi = wmma_bf16(aS[kc].v, bNc, aNi);
      aRh = wmma_bf16(aH[kc].v, bRh, aRh);
      aZh = wmma_bf16(aH[kc].v, bZh, aZh);
      aNh = wmma_bf16(aH[kc].v, bNh, aNh);
    }
    int col = j * 16 + l15;
    float bir = bi[col], biz = bi[128 + col], bin_ = bi[256 + col];
    float bhr = bh[col], bhz = bh[128 + col], bhn = bh[256 + col];
    long long rb = (long long)t * NN + m0 + lh * 8;
#pragma unroll
    for (int v = 0; v < 8; ++v) {
      float* hp = H + (rb + v) * DD + col;
      float hold = *hp;
      float r = sigm(aRi[v] + bir + aRh[v] + bhr);
      float zg = sigm(aZi[v] + biz + aZh[v] + bhz);
      float ng = tanhf(aNi[v] + bin_ + r * (aNh[v] + bhn));
      *hp = (1.0f - zg) * ng + zg * hold;
    }
  }
}

// ---------------- pooling & classifier ----------------

__global__ void k_pool(const float* __restrict__ H, const int* __restrict__ batch,
                       float* __restrict__ sums, float* __restrict__ cnt) {
  int gw = blockIdx.x * 8 + (threadIdx.x >> 5);
  if (gw >= NN) return;
  int lane = threadIdx.x & 31;
  float4 acc; acc.x = acc.y = acc.z = acc.w = 0.0f;
#pragma unroll
  for (int t = 0; t < TT; ++t) {
    float4 v = *(const float4*)(H + ((long long)t * NN + gw) * DD + lane * 4);
    acc.x += v.x; acc.y += v.y; acc.z += v.z; acc.w += v.w;
  }
  int g = batch[gw];
  float* o = sums + (long long)g * DD + lane * 4;
  atomicAdd(o + 0, acc.x); atomicAdd(o + 1, acc.y);
  atomicAdd(o + 2, acc.z); atomicAdd(o + 3, acc.w);
  if (lane == 0) atomicAdd(cnt + g, 1.0f);
}

__global__ void k_cls(const float* __restrict__ sums, const float* __restrict__ cnt,
                      const float* __restrict__ cw, const float* __restrict__ cb,
                      float* __restrict__ out) {
  int i = blockIdx.x * blockDim.x + threadIdx.x;
  if (i >= GG * 2) return;
  int g = i >> 1, c = i & 1;
  float denom = fmaxf(cnt[g], 1.0f);
  float acc = 0.0f;
  for (int d = 0; d < DD; ++d) acc += sums[g * DD + d] * cw[d * 2 + c];
  out[i] = acc / denom + cb[c];
}

// ---------------- launcher ----------------

extern "C" void kernel_launch(void* const* d_in, const int* in_sizes, int n_in,
                              void* d_out, int out_size, void* d_ws, size_t ws_size,
                              hipStream_t stream) {
  (void)in_sizes; (void)n_in; (void)out_size; (void)ws_size;
  const float* x      = (const float*)d_in[0];
  const int*   eidx   = (const int*)d_in[1];
  const int*   eattr  = (const int*)d_in[2];
  const int*   batch  = (const int*)d_in[3];
  const float* lin_w  = (const float*)d_in[4];
  const float* lin_b  = (const float*)d_in[5];
  const float* ggnn_w = (const float*)d_in[6];
  const float* gwih   = (const float*)d_in[7];
  const float* gwhh   = (const float*)d_in[8];
  const float* gbih   = (const float*)d_in[9];
  const float* gbhh   = (const float*)d_in[10];
  const float* cls_w  = (const float*)d_in[11];
  const float* cls_b  = (const float*)d_in[12];

  char* ws = (char*)d_ws;
  size_t off = 0;
  auto grab = [&](size_t bytes) -> void* {
    void* p = ws + off;
    off = (off + bytes + 1023) & ~(size_t)1023;
    return p;
  };
  float*          H    = (float*)grab((size_t)TT * NN * DD * 4);
  float*          S    = (float*)grab((size_t)TT * NN * DD * 4);
  unsigned short* xbf  = (unsigned short*)grab((size_t)NN * DINP * 2);
  unsigned short* lwcm = (unsigned short*)grab((size_t)DD * DINP * 2);
  unsigned short* whhb = (unsigned short*)grab((size_t)TT * 384 * DD * 2);
  unsigned short* wc   = (unsigned short*)grab((size_t)TT * NSTEPS * 384 * DD * 2);
  float*          sums = (float*)grab((size_t)GG * DD * 4);
  float*          cnt  = (float*)grab((size_t)GG * 4);

  // weight / input conversion + Wc precompute
  {
    long long n = (long long)NN * DINP;
    k_cvt_x<<<(int)((n + 255) / 256), 256, 0, stream>>>(x, xbf);
  }
  k_cvt_linw<<<(DD * DINP + 255) / 256, 256, 0, stream>>>(lin_w, lwcm);
  {
    long long n = (long long)TT * 384 * DD;
    k_cvt_bf<<<(int)((n + 255) / 256), 256, 0, stream>>>(gwhh, whhb, n);
  }
  k_wc<<<(TT * NSTEPS * 384 * DD + 255) / 256, 256, 0, stream>>>(ggnn_w, gwih, wc);

  // h0 GEMM, broadcast into all 13 H slices
  k_h0<<<MT * 8 / 8, 256, 0, stream>>>(xbf, lwcm, lin_b, H);

  // 8 GRU propagation steps (all 13 edge types in parallel)
  long long s4 = (long long)TT * NN * DD / 4;
  for (int s = 0; s < NSTEPS; ++s) {
    k_zero<<<(int)((s4 + 255) / 256), 256, 0, stream>>>(S, s4);
    k_scatter<<<EE / 8, 256, 0, stream>>>(H, S, eidx, eattr);
    k_gru<<<(TT * MT + 7) / 8, 256, 0, stream>>>(H, S, wc, whhb, gbih, gbhh, s);
  }

  // mean pool per graph + classifier
  k_zero<<<(GG * DD / 4 + 255) / 256, 256, 0, stream>>>(sums, GG * DD / 4);
  k_zero<<<1, 64, 0, stream>>>(cnt, GG / 4);
  k_pool<<<(NN + 7) / 8, 256, 0, stream>>>(H, batch, sums, cnt);
  k_cls<<<2, 256, 0, stream>>>(sums, cnt, cls_w, cls_b, (float*)d_out);
}
